// BidirectionalAttention_18545668784918
// MI455X (gfx1250) — matrix operand order repaired and checked by reading, not compiled, not run on hardware
//
#include <hip/hip_runtime.h>
#include <math.h>

#define B_    2
#define S_    4096
#define D_    768
#define H_    6
#define HKV_  2
#define HD_   128
#define REP_  3
#define MROWS (B_ * S_)              // 8192
#define QSCALE 0.08838834764831845f  // 1/sqrt(128)

typedef __bf16 v16bf __attribute__((ext_vector_type(16)));
typedef __bf16 v8bf  __attribute__((ext_vector_type(8)));
typedef float  v8f   __attribute__((ext_vector_type(8)));
typedef int    v4i   __attribute__((vector_size(16)));

union Frag {
    v16bf v;
    v8bf  h[2];
};

static __device__ inline v8f wmma_bf16(const Frag& a, const Frag& b, v8f c) {
    return __builtin_amdgcn_wmma_f32_16x16x32_bf16(
        /*neg_a=*/false, a.v, /*neg_b=*/false, b.v,
        /*c_mod=*/(short)0, c, /*reuse_a=*/false, /*reuse_b=*/false);
}

// ---- async global->LDS copy (CDNA5 GLOBAL_LOAD_ASYNC_TO_LDS_B128, ASYNCcnt)
#if defined(__HIP_DEVICE_COMPILE__) && __has_builtin(__builtin_amdgcn_global_load_async_to_lds_b128)
#define HAVE_ASYNC_LDS 1
#endif

static __device__ inline void async_cp16(__bf16* dst_lds, const __bf16* src_glb) {
#ifdef HAVE_ASYNC_LDS
    __builtin_amdgcn_global_load_async_to_lds_b128(
        (__attribute__((address_space(1))) v4i*)(src_glb),
        (__attribute__((address_space(3))) v4i*)(dst_lds),
        /*imm offset=*/0, /*cpol=*/0);
#else
    *(v8bf*)dst_lds = *(const v8bf*)src_glb;  // synchronous fallback via VGPR
#endif
}

static __device__ inline void async_wait0() {
#ifdef HAVE_ASYNC_LDS
#if __has_builtin(__builtin_amdgcn_s_wait_asynccnt)
    __builtin_amdgcn_s_wait_asynccnt(0);
#else
    asm volatile("s_wait_asynccnt 0x0" ::: "memory");
#endif
#endif
}

// ---------------------------------------------------------------- conversions
__global__ void cvt_f32_bf16(const float* __restrict__ s, __bf16* __restrict__ d, int n) {
    int i = blockIdx.x * blockDim.x + threadIdx.x;
    if (i < n) d[i] = (__bf16)s[i];
}

// w[K][N] (row-major f32) -> wt[N][K] (row-major bf16)
__global__ void transpose_cvt(const float* __restrict__ w, __bf16* __restrict__ wt, int K, int N) {
    int i = blockIdx.x * blockDim.x + threadIdx.x;
    if (i >= K * N) return;
    int kk = i / N, n = i - kk * N;
    wt[(size_t)n * K + kk] = (__bf16)w[i];
}

// ---------------------------------------------------------------- bf16 GEMM
// C[M,N] = A[M,K] (bf16 row-major) x WT[N,K] (bf16, pre-transposed weight)
// block = 128 threads = 4 waves; each wave computes a 16x64 tile.
// mode 0: bf16 out row-major [M,N] (scaled)      (q / k)
// mode 1: bf16 out scattered as vT[b][kv][d][s]  (v)
// mode 2: f32 out row-major [M,N]                (final projection)
__global__ __launch_bounds__(128) void gemm_bf16_k(
    const __bf16* __restrict__ A, const __bf16* __restrict__ WT, void* __restrict__ out,
    int M, int N, int K, float scale, int mode) {
    int w    = threadIdx.x >> 5;
    int ln   = threadIdx.x & 31;
    int half = ln >> 4, lh = ln & 15;
    int m0 = blockIdx.x * 64 + w * 16;
    int n0 = blockIdx.y * 64;

    v8f vz = {0.f, 0.f, 0.f, 0.f, 0.f, 0.f, 0.f, 0.f};
    v8f acc[4] = {vz, vz, vz, vz};

    const __bf16* arow = A + (size_t)(m0 + lh) * K;
    for (int k0 = 0; k0 < K; k0 += 32) {
        Frag af;
        const __bf16* ap = arow + k0 + half * 16;
        af.h[0] = *(const v8bf*)ap;
        af.h[1] = *(const v8bf*)(ap + 8);
#pragma unroll
        for (int cc = 0; cc < 4; ++cc) {
            Frag bfr;
            bfr.v = *(const v16bf*)(WT + (size_t)(n0 + cc * 16 + lh) * K + k0 + half * 16);
            acc[cc] = wmma_bf16(af, bfr, acc[cc]);
        }
    }

#pragma unroll
    for (int cc = 0; cc < 4; ++cc) {
#pragma unroll
        for (int r = 0; r < 8; ++r) {
            int mm = m0 + half * 8 + r;     // D layout: row = 8*(lane/16) + vgpr
            int nn = n0 + cc * 16 + lh;     //           col = lane%16
            float val = acc[cc][r] * scale;
            if (mode == 0) {
                ((__bf16*)out)[(size_t)mm * N + nn] = (__bf16)val;
            } else if (mode == 1) {
                int bb = mm / S_, s = mm - bb * S_;
                int hv = nn / HD_, d = nn - hv * HD_;
                ((__bf16*)out)[(((size_t)bb * HKV_ + hv) * HD_ + d) * S_ + s] = (__bf16)val;
            } else {
                ((float*)out)[(size_t)mm * N + nn] = val;
            }
        }
    }
}

// ---------------------------------------------------------------- flash attention
// q  : bf16 [B*S][H*HD]      (already scaled by 1/sqrt(HD))
// k  : bf16 [B*S][HKV*HD]
// vt : bf16 [B][HKV][HD][S]  (transposed V)
// ao : bf16 [B*S][H*HD]
// grid (S/64, H, B), block 128 = 4 waves, each wave one 16-query tile.
// The 4 waves share (b,h), so K/V 32-key tiles are staged ONCE per block into
// LDS via async global->LDS copies, then all waves read fragments from LDS.
#define KT_LD 136   // 128 + 8 pad  -> row stride 272B (16B-aligned, bank-spread)
#define VT_LD 40    //  32 + 8 pad  -> row stride  80B (16B-aligned, bank-spread)
#define PT_LD 40    //  32 + 8 pad

__global__ __launch_bounds__(128) void attn_fwd(
    const __bf16* __restrict__ q, const __bf16* __restrict__ k,
    const __bf16* __restrict__ vt, __bf16* __restrict__ ao) {
    __shared__ __bf16 Kt[32][KT_LD];     // key-major:  [key][dim]
    __shared__ __bf16 Vt[128][VT_LD];    // dim-major:  [dim][key]
    __shared__ __bf16 P[4][16][PT_LD];   // per-wave probability tile

    int tid  = threadIdx.x;
    int w    = tid >> 5;
    int ln   = tid & 31;
    int half = ln >> 4, lh = ln & 15;
    int qt = blockIdx.x * 4 + w;
    int h  = blockIdx.y;
    int b  = blockIdx.z;
    int hkv = h / REP_;                  // repeat_interleave: head h -> kv head h/3
    int s0 = qt * 16;

    // Q A-fragments for the full 128-dim (4 x K=32 chunks), loaded once.
    Frag qf[4];
    const __bf16* qrow = q + (size_t)(b * S_ + s0 + lh) * (H_ * HD_) + h * HD_;
#pragma unroll
    for (int dk = 0; dk < 4; ++dk) {
        const __bf16* p0 = qrow + dk * 32 + half * 16;
        qf[dk].h[0] = *(const v8bf*)p0;
        qf[dk].h[1] = *(const v8bf*)(p0 + 8);
    }

    v8f vz = {0.f, 0.f, 0.f, 0.f, 0.f, 0.f, 0.f, 0.f};
    v8f acc[8] = {vz, vz, vz, vz, vz, vz, vz, vz};  // O tile: 16 x 128 f32
    float m_i[8], l_i[8];
#pragma unroll
    for (int r = 0; r < 8; ++r) { m_i[r] = -1e30f; l_i[r] = 0.f; }

    const __bf16* kbase = k + (size_t)(b * S_) * (HKV_ * HD_) + hkv * HD_;
    const __bf16* vbase = vt + ((size_t)b * HKV_ + hkv) * HD_ * S_;

    for (int n0 = 0; n0 < S_; n0 += 32) {
        // ---- stage K tile (32 keys x 128 dims) and V tile (128 dims x 32 keys)
        // 16B chunks, 128 threads x 4 passes each => ASYNCcnt-tracked copies
#pragma unroll
        for (int j = 0; j < 4; ++j) {
            int c = j * 128 + tid;           // 0..511, 16 chunks per key row
            int key = c >> 4, col = c & 15;
            async_cp16(&Kt[key][col * 8],
                       kbase + (size_t)(n0 + key) * (HKV_ * HD_) + col * 8);
        }
#pragma unroll
        for (int j = 0; j < 4; ++j) {
            int c = j * 128 + tid;           // 0..511, 4 chunks per dim row
            int d = c >> 2, kg = c & 3;
            async_cp16(&Vt[d][kg * 8],
                       vbase + (size_t)d * S_ + n0 + kg * 8);
        }
        async_wait0();
        __syncthreads();                      // K/V tiles visible to all waves

        // ---- scores: two 16-key sub-tiles, K-dim reduction of 128 (4 WMMAs each)
        v8f s0a = vz, s1a = vz;
#pragma unroll
        for (int dk = 0; dk < 4; ++dk) {
            Frag b0, b1;
            b0.h[0] = *(const v8bf*)&Kt[lh][dk * 32 + half * 16];
            b0.h[1] = *(const v8bf*)&Kt[lh][dk * 32 + half * 16 + 8];
            b1.h[0] = *(const v8bf*)&Kt[16 + lh][dk * 32 + half * 16];
            b1.h[1] = *(const v8bf*)&Kt[16 + lh][dk * 32 + half * 16 + 8];
            s0a = wmma_bf16(qf[dk], b0, s0a);
            s1a = wmma_bf16(qf[dk], b1, s1a);
        }

        // ---- online softmax update (row-wise across the 16-lane group)
        float alpha[8];
#pragma unroll
        for (int r = 0; r < 8; ++r) {
            float s = fmaxf(s0a[r], s1a[r]);
            s = fmaxf(s, __shfl_xor(s, 8, 16));
            s = fmaxf(s, __shfl_xor(s, 4, 16));
            s = fmaxf(s, __shfl_xor(s, 2, 16));
            s = fmaxf(s, __shfl_xor(s, 1, 16));
            float mn = fmaxf(m_i[r], s);
            float a  = __expf(m_i[r] - mn);
            m_i[r]   = mn;
            float p0 = __expf(s0a[r] - mn);
            float p1 = __expf(s1a[r] - mn);
            float rs = p0 + p1;
            rs += __shfl_xor(rs, 8, 16);
            rs += __shfl_xor(rs, 4, 16);
            rs += __shfl_xor(rs, 2, 16);
            rs += __shfl_xor(rs, 1, 16);
            l_i[r] = l_i[r] * a + rs;
            alpha[r] = a;
            // stage P (16x32) to LDS to re-layout D-fragment -> A-fragment
            P[w][half * 8 + r][lh]      = (__bf16)p0;
            P[w][half * 8 + r][16 + lh] = (__bf16)p1;
        }
#pragma unroll
        for (int c = 0; c < 8; ++c)
#pragma unroll
            for (int r = 0; r < 8; ++r) acc[c][r] *= alpha[r];

        // per-wave LDS ops are in-order (DScnt): P write->read needs no barrier
        Frag pa;
        pa.h[0] = *(const v8bf*)&P[w][lh][half * 16];
        pa.h[1] = *(const v8bf*)&P[w][lh][half * 16 + 8];

        // ---- O += P x V  (8 dim-chunks of 16, K=32 keys)
#pragma unroll
        for (int c = 0; c < 8; ++c) {
            Frag vbf;
            vbf.h[0] = *(const v8bf*)&Vt[c * 16 + lh][half * 16];
            vbf.h[1] = *(const v8bf*)&Vt[c * 16 + lh][half * 16 + 8];
            acc[c] = wmma_bf16(pa, vbf, acc[c]);
        }
        __syncthreads();   // all waves done reading Kt/Vt before next stage
    }

    // ---- normalize and write attention output
#pragma unroll
    for (int r = 0; r < 8; ++r) {
        float inv = 1.0f / l_i[r];
        __bf16* orow = ao + (size_t)(b * S_ + s0 + half * 8 + r) * (H_ * HD_) + h * HD_;
#pragma unroll
        for (int c = 0; c < 8; ++c) orow[c * 16 + lh] = (__bf16)(acc[c][r] * inv);
    }
}

// ---------------------------------------------------------------- launch
extern "C" void kernel_launch(void* const* d_in, const int* in_sizes, int n_in,
                              void* d_out, int out_size, void* d_ws, size_t ws_size,
                              hipStream_t stream) {
    (void)in_sizes; (void)n_in; (void)out_size; (void)ws_size;
    const float* x  = (const float*)d_in[0];
    const float* wq = (const float*)d_in[1];
    const float* wk = (const float*)d_in[2];
    const float* wv = (const float*)d_in[3];
    const float* wo = (const float*)d_in[4];
    float* out = (float*)d_out;

    char* p = (char*)d_ws;
    auto take = [&](size_t bytes) -> char* {
        char* r = p;
        p += (bytes + 255) & ~(size_t)255;
        return r;
    };
    __bf16* xb  = (__bf16*)take((size_t)MROWS * D_ * 2);
    __bf16* qb  = (__bf16*)take((size_t)MROWS * H_ * HD_ * 2);
    __bf16* kb  = (__bf16*)take((size_t)MROWS * HKV_ * HD_ * 2);
    __bf16* vb  = (__bf16*)take((size_t)MROWS * HKV_ * HD_ * 2);
    __bf16* ao  = (__bf16*)take((size_t)MROWS * H_ * HD_ * 2);
    __bf16* wqT = (__bf16*)take((size_t)D_ * H_ * HD_ * 2);
    __bf16* wkT = (__bf16*)take((size_t)D_ * HKV_ * HD_ * 2);
    __bf16* wvT = (__bf16*)take((size_t)D_ * HKV_ * HD_ * 2);
    __bf16* woT = (__bf16*)take((size_t)H_ * HD_ * D_ * 2);

    int nx = MROWS * D_;
    cvt_f32_bf16<<<(nx + 255) / 256, 256, 0, stream>>>(x, xb, nx);
    transpose_cvt<<<(D_ * (H_ * HD_) + 255) / 256, 256, 0, stream>>>(wq, wqT, D_, H_ * HD_);
    transpose_cvt<<<(D_ * (HKV_ * HD_) + 255) / 256, 256, 0, stream>>>(wk, wkT, D_, HKV_ * HD_);
    transpose_cvt<<<(D_ * (HKV_ * HD_) + 255) / 256, 256, 0, stream>>>(wv, wvT, D_, HKV_ * HD_);
    transpose_cvt<<<((H_ * HD_) * D_ + 255) / 256, 256, 0, stream>>>(wo, woT, H_ * HD_, D_);

    // projections (scale folded into q)
    gemm_bf16_k<<<dim3(MROWS / 64, (H_ * HD_) / 64), 128, 0, stream>>>(
        xb, wqT, qb, MROWS, H_ * HD_, D_, QSCALE, 0);
    gemm_bf16_k<<<dim3(MROWS / 64, (HKV_ * HD_) / 64), 128, 0, stream>>>(
        xb, wkT, kb, MROWS, HKV_ * HD_, D_, 1.0f, 0);
    gemm_bf16_k<<<dim3(MROWS / 64, (HKV_ * HD_) / 64), 128, 0, stream>>>(
        xb, wvT, vb, MROWS, HKV_ * HD_, D_, 1.0f, 1);

    // streaming (flash) bidirectional attention
    attn_fwd<<<dim3(S_ / 64, H_, B_), 128, 0, stream>>>(qb, kb, vb, ao);

    // output projection, f32 into d_out
    gemm_bf16_k<<<dim3(MROWS / 64, D_ / 64), 128, 0, stream>>>(
        ao, woT, out, MROWS, D_, H_ * HD_, 1.0f, 2);
}
